// GCN_979252543624
// MI455X (gfx1250) — matrix operand order repaired and checked by reading, not compiled, not run on hardware
//
#include <hip/hip_runtime.h>
#include <math.h>

// GCN forward for MI455X (gfx1250, wave32).
// Memory-bound: ~2.7 GB gather/scatter traffic dominates (~120us at 23.3 TB/s).
// GEMMs use v_wmma_f32_16x16x32_bf16 with W pre-swizzled into fragment-major LDS
// (one 32B ds_load per B fragment) and v_perm-packed bf16 A fragments.

#define N_NODES 100000
#define F_IN    128
#define HID     64
#define CLS     40
#define NPAD    48   // CLS padded to 3x16 tiles

typedef __attribute__((ext_vector_type(16))) __bf16 v16bf;
typedef __attribute__((ext_vector_type(8)))  float  v8f;

__device__ __forceinline__ unsigned short f2bfbits(float f) {
    union { float f; unsigned u; } x; x.f = f;
    return (unsigned short)((x.u + 0x7FFFu + ((x.u >> 16) & 1u)) >> 16); // RNE
}

// Pack two floats to two bf16 (round-half-up) with a single v_perm_b32.
__device__ __forceinline__ unsigned pack2bf(float lo, float hi) {
    union { float f; unsigned u; } a, b; a.f = lo; b.f = hi;
    unsigned x = a.u + 0x8000u;
    unsigned y = b.u + 0x8000u;
    return __builtin_amdgcn_perm(y, x, 0x07060302u); // {y[31:16], x[31:16]}
}

// ---------------- degree / norm ----------------

__global__ __launch_bounds__(256) void k_deg_init(float* __restrict__ deg) {
    int i = blockIdx.x * 256 + threadIdx.x;
    if (i < N_NODES) deg[i] = 1.0f;                 // self loop contribution
}

__global__ __launch_bounds__(256) void k_deg_edges(const int* __restrict__ dst, int E,
                                                   float* __restrict__ deg) {
    int e = blockIdx.x * 256 + threadIdx.x;
    if (e < E) atomicAdd(&deg[dst[e]], 1.0f);
}

__global__ __launch_bounds__(256) void k_rsqrt(float* __restrict__ deg) {
    int i = blockIdx.x * 256 + threadIdx.x;
    if (i < N_NODES) deg[i] = rsqrtf(deg[i]);       // deg >= 1 always
}

// ---------------- bias broadcast init ----------------

template<int F>
__global__ __launch_bounds__(256) void k_bias(const float* __restrict__ b,
                                              float* __restrict__ out) {
    long long idx = (long long)blockIdx.x * 256 + threadIdx.x;
    if (idx < (long long)N_NODES * F) out[idx] = b[(int)(idx % F)];
}

// ---------------- edge aggregation (scatter-add) ----------------
// One thread per (edge, feature): coalesced row gather + coalesced
// global_atomic_add_f32 per destination row (atomics resolve in L2).

template<int F>
__global__ __launch_bounds__(256) void k_agg(const float* __restrict__ HW,
                                             const float* __restrict__ dinv,
                                             const int* __restrict__ src,
                                             const int* __restrict__ dst,
                                             int E, float* __restrict__ AGG) {
    long long idx = (long long)blockIdx.x * 256 + threadIdx.x;
    long long total = (long long)(E + N_NODES) * F;
    if (idx >= total) return;
    int f = (int)(idx % F);
    int e = (int)(idx / F);
    int s, d;
    if (e < E) { s = src[e]; d = dst[e]; }
    else       { s = d = e - E; }                   // appended self loop
    float nrm = dinv[s] * dinv[d];
    atomicAdd(&AGG[(size_t)d * F + f], HW[(size_t)s * F + f] * nrm);
}

// ---------------- A fragment: 16x32 bf16 from a row-major f32 row ----------------
// lane<16 -> M=lane&15, K = {kb..kb+7, kb+16..kb+23} with kb = K0 + (lane>>4)*8.
template<bool RELU>
__device__ __forceinline__ v16bf load_a_frag(const float* __restrict__ rowp, int kb) {
    float t[16];
    float4 p0 = *(const float4*)(rowp + kb);
    float4 p1 = *(const float4*)(rowp + kb + 4);
    float4 p2 = *(const float4*)(rowp + kb + 16);
    float4 p3 = *(const float4*)(rowp + kb + 20);
    t[0]=p0.x;  t[1]=p0.y;  t[2]=p0.z;  t[3]=p0.w;
    t[4]=p1.x;  t[5]=p1.y;  t[6]=p1.z;  t[7]=p1.w;
    t[8]=p2.x;  t[9]=p2.y;  t[10]=p2.z; t[11]=p2.w;
    t[12]=p3.x; t[13]=p3.y; t[14]=p3.z; t[15]=p3.w;
    union { v16bf v; unsigned u[8]; } A;
#pragma unroll
    for (int j = 0; j < 8; ++j) {
        float lo = t[2*j], hi = t[2*j+1];
        if (RELU) { lo = fmaxf(lo, 0.0f); hi = fmaxf(hi, 0.0f); }
        A.u[j] = pack2bf(lo, hi);
    }
    return A.v;
}

// ---------------- GEMM1: hw1 = x @ W1   [100000x128]@[128x64] ----------------
// W1 pre-swizzled to fragment-major LDS: Wfrag[(ks*4+nt)*32 + lane] is lane's
// whole 16-element B fragment (32B, one aligned LDS vector load).

__global__ __launch_bounds__(256) void k_gemm1(const float* __restrict__ X,
                                               const float* __restrict__ W1,
                                               float* __restrict__ HW) {
    __shared__ v16bf Wfrag[4 * 4 * 32];             // 16 KB
    {
        unsigned short* Wf = (unsigned short*)Wfrag;
        for (int j = threadIdx.x; j < 4 * 4 * 32 * 16; j += 256) {
            int i    = j & 15;
            int lane = (j >> 4) & 31;
            int nt   = (j >> 9) & 3;
            int ks   = j >> 11;
            int k    = ks * 32 + (lane >> 4) * 8 + (i < 8 ? i : i + 8);   // B: K = half*8 + i (i<8), 16 + half*8 + (i-8) (i>=8)
            int col  = nt * 16 + (lane & 15);
            Wf[j] = f2bfbits(W1[k * HID + col]);
        }
    }
    __syncthreads();

    int tile = blockIdx.x * 8 + (threadIdx.x >> 5);      // 32 rows per wave
    if (tile >= N_NODES / 32) return;                    // wave-uniform exit
    int lane = threadIdx.x & 31;
    int half = lane >> 4, mr = lane & 15;
    const float* xr0 = X + (size_t)(tile * 32 + mr) * F_IN;
    const float* xr1 = xr0 + 16 * F_IN;

    v8f z = {0,0,0,0,0,0,0,0};
    v8f acc[2][4];
#pragma unroll
    for (int m = 0; m < 2; ++m)
#pragma unroll
        for (int nt = 0; nt < 4; ++nt) acc[m][nt] = z;

#pragma unroll
    for (int ks = 0; ks < 4; ++ks) {                     // K = 128 step 32
        int kb = ks * 32 + half * 8;
        v16bf A0 = load_a_frag<false>(xr0, kb);
        v16bf A1 = load_a_frag<false>(xr1, kb);
#pragma unroll
        for (int nt = 0; nt < 4; ++nt) {                 // N = 64, tiles of 16
            v16bf B = Wfrag[(ks * 4 + nt) * 32 + lane];  // 2x ds_load_b128
            acc[0][nt] = __builtin_amdgcn_wmma_f32_16x16x32_bf16(
                false, A0, false, B, (short)0, acc[0][nt], false, false);
            acc[1][nt] = __builtin_amdgcn_wmma_f32_16x16x32_bf16(
                false, A1, false, B, (short)0, acc[1][nt], false, false);
        }
    }
#pragma unroll
    for (int m = 0; m < 2; ++m) {
#pragma unroll
        for (int nt = 0; nt < 4; ++nt) {
            int col = nt * 16 + mr;
#pragma unroll
            for (int i = 0; i < 8; ++i) {
                int row = tile * 32 + m * 16 + half * 8 + i;
                HW[(size_t)row * HID + col] = acc[m][nt][i];
            }
        }
    }
}

// ------- GEMM2: hw2 = relu(agg1) @ W2  [100000x64]@[64x40] (N padded to 48) -------

__global__ __launch_bounds__(256) void k_gemm2(const float* __restrict__ Hin,
                                               const float* __restrict__ W2,
                                               float* __restrict__ HW2) {
    __shared__ v16bf Wfrag[2 * 3 * 32];             // 6 KB, zero-padded cols 40..47
    {
        unsigned short* Wf = (unsigned short*)Wfrag;
        for (int j = threadIdx.x; j < 2 * 3 * 32 * 16; j += 256) {
            int i    = j & 15;
            int lane = (j >> 4) & 31;
            int nt   = (j >> 9) % 3;
            int ks   = j / (3 * 32 * 16);
            int k    = ks * 32 + (lane >> 4) * 8 + (i < 8 ? i : i + 8);
            int col  = nt * 16 + (lane & 15);
            Wf[j] = (col < CLS) ? f2bfbits(W2[k * CLS + col]) : (unsigned short)0;
        }
    }
    __syncthreads();

    int tile = blockIdx.x * 8 + (threadIdx.x >> 5);      // 32 rows per wave
    if (tile >= N_NODES / 32) return;
    int lane = threadIdx.x & 31;
    int half = lane >> 4, mr = lane & 15;
    const float* hr0 = Hin + (size_t)(tile * 32 + mr) * HID;
    const float* hr1 = hr0 + 16 * HID;

    v8f z = {0,0,0,0,0,0,0,0};
    v8f acc[2][3];
#pragma unroll
    for (int m = 0; m < 2; ++m)
#pragma unroll
        for (int nt = 0; nt < 3; ++nt) acc[m][nt] = z;

#pragma unroll
    for (int ks = 0; ks < 2; ++ks) {                     // K = 64 step 32
        int kb = ks * 32 + half * 8;
        v16bf A0 = load_a_frag<true>(hr0, kb);           // fused ReLU
        v16bf A1 = load_a_frag<true>(hr1, kb);
#pragma unroll
        for (int nt = 0; nt < 3; ++nt) {
            v16bf B = Wfrag[(ks * 3 + nt) * 32 + lane];
            acc[0][nt] = __builtin_amdgcn_wmma_f32_16x16x32_bf16(
                false, A0, false, B, (short)0, acc[0][nt], false, false);
            acc[1][nt] = __builtin_amdgcn_wmma_f32_16x16x32_bf16(
                false, A1, false, B, (short)0, acc[1][nt], false, false);
        }
    }
#pragma unroll
    for (int m = 0; m < 2; ++m) {
#pragma unroll
        for (int nt = 0; nt < 3; ++nt) {
            int col = nt * 16 + mr;
            if (col < CLS) {
#pragma unroll
                for (int i = 0; i < 8; ++i) {
                    int row = tile * 32 + m * 16 + half * 8 + i;
                    HW2[(size_t)row * CLS + col] = acc[m][nt][i];
                }
            }
        }
    }
}

// ---------------- log_softmax, one wave (32 lanes) per 40-wide row ----------------

__global__ __launch_bounds__(256) void k_logsoftmax(float* __restrict__ out) {
    int row = blockIdx.x * 8 + (threadIdx.x >> 5);
    if (row >= N_NODES) return;
    int lane = threadIdx.x & 31;
    float* r = out + (size_t)row * CLS;
    float v0 = (lane < CLS)      ? r[lane]      : -INFINITY;
    float v1 = (lane + 32 < CLS) ? r[lane + 32] : -INFINITY;
    float m = fmaxf(v0, v1);
#pragma unroll
    for (int o = 16; o > 0; o >>= 1) m = fmaxf(m, __shfl_xor(m, o, 32));
    float s = 0.0f;
    if (lane < CLS)      s += expf(v0 - m);
    if (lane + 32 < CLS) s += expf(v1 - m);
#pragma unroll
    for (int o = 16; o > 0; o >>= 1) s += __shfl_xor(s, o, 32);
    float lse = m + logf(s);
    if (lane < CLS)      r[lane]      = v0 - lse;
    if (lane + 32 < CLS) r[lane + 32] = v1 - lse;
}

// ---------------- launcher ----------------

extern "C" void kernel_launch(void* const* d_in, const int* in_sizes, int n_in,
                              void* d_out, int out_size, void* d_ws, size_t ws_size,
                              hipStream_t stream) {
    const float* x  = (const float*)d_in[0];
    const int*   ei = (const int*)  d_in[1];
    const float* W1 = (const float*)d_in[2];
    const float* b1 = (const float*)d_in[3];
    const float* W2 = (const float*)d_in[4];
    const float* b2 = (const float*)d_in[5];
    float* out = (float*)d_out;

    int E = in_sizes[1] / 2;
    const int* src = ei;
    const int* dst = ei + E;

    // scratch layout: dinv[N] | buf1[N*64] (hw1, then hw2) | agg1[N*64]
    float* dinv = (float*)d_ws;
    float* buf1 = dinv + N_NODES;
    float* agg1 = buf1 + (size_t)N_NODES * HID;

    const int TB = 256;
    unsigned gN    = (N_NODES + TB - 1) / TB;
    unsigned gE    = (E + TB - 1) / TB;
    unsigned gGemm = ((N_NODES / 32) + 7) / 8;        // 8 waves x 32 rows per block

    // 1) symmetric normalization coefficients
    k_deg_init <<<gN, TB, 0, stream>>>(dinv);
    k_deg_edges<<<gE, TB, 0, stream>>>(dst, E, dinv);
    k_rsqrt    <<<gN, TB, 0, stream>>>(dinv);

    // 2) layer 1: hw1 = x@W1 ; agg1 = b1 + scatter-add(norm * hw1[src])
    k_gemm1<<<gGemm, TB, 0, stream>>>(x, W1, buf1);
    long long n1 = (long long)N_NODES * HID;
    k_bias<HID><<<(unsigned)((n1 + TB - 1) / TB), TB, 0, stream>>>(b1, agg1);
    long long t1 = (long long)(E + N_NODES) * HID;
    k_agg<HID><<<(unsigned)((t1 + TB - 1) / TB), TB, 0, stream>>>(buf1, dinv, src, dst, E, agg1);

    // 3) layer 2: hw2 = relu(agg1)@W2 ; out = b2 + scatter-add(norm * hw2[src])
    k_gemm2<<<gGemm, TB, 0, stream>>>(agg1, W2, buf1);
    long long n2 = (long long)N_NODES * CLS;
    k_bias<CLS><<<(unsigned)((n2 + TB - 1) / TB), TB, 0, stream>>>(b2, out);
    long long t2 = (long long)(E + N_NODES) * CLS;
    k_agg<CLS><<<(unsigned)((t2 + TB - 1) / TB), TB, 0, stream>>>(buf1, dinv, src, dst, E, out);

    // 4) row-wise log_softmax in place
    k_logsoftmax<<<(N_NODES + 7) / 8, TB, 0, stream>>>(out);
}
